// CNNBCHEncoder_12893491822746
// MI455X (gfx1250) — 8-bit matrix operand order repaired and checked by reading, compiled, not run on hardware
//
#include <hip/hip_runtime.h>
#include <stdint.h>

typedef __attribute__((ext_vector_type(8))) int v8i;

#define MSG_BITS 1000
#define PAR_BITS 256
#define OUT_BITS 1256
#define KC_COUNT 16
#define NT_COUNT 16
#define WAVES_PER_BLOCK 4
#define LDS_DW_PER_WAVE 4096

__global__ void pack_B_kernel(const float* __restrict__ Gp,
                              uint32_t* __restrict__ wsB) {
    int t    = blockIdx.x * blockDim.x + threadIdx.x;
    int v    = t & 7;
    int lane = (t >> 3) & 31;
    int kc   = (t >> 8) & 15;
    int nt   = t >> 12;
    int n     = nt * 16 + (lane & 15);
    int kbase = kc * 64 + ((lane >> 4) & 1) * 16 + ((v >= 4) ? 32 : 0) + (v & 3) * 4;
    uint32_t w = 0u;
#pragma unroll
    for (int b = 0; b < 4; ++b) {
        int k = kbase + b;
        uint32_t bit = (k < MSG_BITS) ? (uint32_t)Gp[(size_t)k * PAR_BITS + n] : 0u;
        w |= (bit & 1u) << (8 * b);
    }
    wsB[(size_t)((nt * 16 + kc) * 32 + lane) * 8 + v] = w;
}

__global__ void __launch_bounds__(WAVES_PER_BLOCK * 32)
rs_encode_kernel(const float* __restrict__ msg,
                 const uint32_t* __restrict__ wsB,
                 float* __restrict__ out,
                 int mTiles) {
    __shared__ uint32_t ldsA[WAVES_PER_BLOCK * LDS_DW_PER_WAVE];

    const int lane  = threadIdx.x & 31;
    const int wave  = threadIdx.x >> 5;
    const int mTile = blockIdx.x * WAVES_PER_BLOCK + wave;
    if (mTile >= mTiles) return;
    const int mBase = mTile * 16;
    const int wbase = wave * LDS_DW_PER_WAVE;

    const float* mrow = msg + (size_t)mBase * MSG_BITS;
    float*       orow = out + (size_t)mBase * OUT_BITS;

    for (int i = lane; i < 16 * 250; i += 32) {
        int r = i / 250;
        int c = i - r * 250;
        float4 q = ((const float4*)(mrow + (size_t)r * MSG_BITS))[c];
        ((float4*)(orow + (size_t)r * OUT_BITS))[c] = q;
    }

    const int halfK = (lane >> 4) & 1;
    const int m     = lane & 15;
    const float* arow = mrow + (size_t)m * MSG_BITS;

#pragma unroll
    for (int kc = 0; kc < KC_COUNT; ++kc) {
        uint32_t wv[8];
#pragma unroll
        for (int v = 0; v < 8; ++v) {
            int kb = kc * 64 + halfK * 16 + (v >> 2) * 32 + (v & 3) * 4;
            uint32_t w = 0u;
            if (kb < MSG_BITS) {
                float4 f = *(const float4*)(arow + kb);
                w = ((uint32_t)f.x & 1u)
                  | (((uint32_t)f.y & 1u) << 8)
                  | (((uint32_t)f.z & 1u) << 16)
                  | (((uint32_t)f.w & 1u) << 24);
            }
            wv[v] = w;
        }
        uint32_t* dst = &ldsA[wbase + kc * 256 + lane * 8];
        *(uint4*)(dst + 0) = make_uint4(wv[0], wv[1], wv[2], wv[3]);
        *(uint4*)(dst + 4) = make_uint4(wv[4], wv[5], wv[6], wv[7]);
    }

    const int colBase = MSG_BITS + (lane & 15);
    const int rbase   = halfK * 8;

    for (int ntp = 0; ntp < NT_COUNT / 2; ++ntp) {
        const int nt0 = ntp * 2;
        v8i acc0 = {};
        v8i acc1 = {};
        const uint32_t* bp0 = wsB + (size_t)(nt0 * KC_COUNT) * 256;
        const uint32_t* bp1 = bp0 + (size_t)KC_COUNT * 256;
#pragma unroll
        for (int kk = 0; kk < KC_COUNT; ++kk) {
            int kc = (kk + ntp) & (KC_COUNT - 1);
            v8i a  = *(const v8i*)&ldsA[wbase + kc * 256 + lane * 8];
            v8i b0 = *(const v8i*)(bp0 + (size_t)(kc * 32 + lane) * 8);
            v8i b1 = *(const v8i*)(bp1 + (size_t)(kc * 32 + lane) * 8);
            acc0 = __builtin_amdgcn_wmma_i32_16x16x64_iu8(false, a, false, b0,
                                                          acc0, false, false);
            acc1 = __builtin_amdgcn_wmma_i32_16x16x64_iu8(false, a, false, b1,
                                                          acc1, false, false);
        }
#pragma unroll
        for (int r = 0; r < 8; ++r) {
            size_t rowOff = (size_t)(rbase + r) * OUT_BITS;
            orow[rowOff + colBase + nt0 * 16]      = (float)(acc0[r] & 1);
            orow[rowOff + colBase + nt0 * 16 + 16] = (float)(acc1[r] & 1);
        }
    }
}

extern "C" void kernel_launch(void* const* d_in, const int* in_sizes, int n_in,
                              void* d_out, int out_size, void* d_ws, size_t ws_size,
                              hipStream_t stream) {
    const float* msg = (const float*)d_in[0];
    const float* Gp  = (const float*)d_in[1];
    float*       out = (float*)d_out;
    uint32_t*    wsB = (uint32_t*)d_ws;

    const int batch  = in_sizes[0] / MSG_BITS;
    const int mTiles = (batch + 15) / 16;

    pack_B_kernel<<<65536 / 256, 256, 0, stream>>>(Gp, wsB);

    rs_encode_kernel<<<(mTiles + WAVES_PER_BLOCK - 1) / WAVES_PER_BLOCK,
                       WAVES_PER_BLOCK * 32, 0, stream>>>(msg, wsB, out, mTiles);
}
